// GATModel_67242007986234
// MI455X (gfx1250) — matrix operand order repaired and checked by reading, not compiled, not run on hardware
//
#include <hip/hip_runtime.h>

// ---------------------------------------------------------------------------
// Fused GATv2 forward for MI455X (gfx1250), wave32, WMMA bf16 path.
// Only layer l = L-1 = 2 matters (reference feeds x0 into every layer).
// Round 2: stage x[b] into LDS via GLOBAL_LOAD_ASYNC_TO_LDS_B128 (ASYNCcnt).
// ---------------------------------------------------------------------------

typedef __attribute__((ext_vector_type(16))) __bf16 v16bf;
typedef __attribute__((ext_vector_type(8)))  __bf16 v8bf;
typedef __attribute__((ext_vector_type(8)))  float  v8f;
typedef __bf16 bf16;

// workspace layout (bytes)
#define WS_WEXPT 0          // 256*64  bf16 = 32768
#define WS_WLT   32768      // 256*256 bf16 = 131072
#define WS_WRT   163840     // 256*256 bf16 = 131072
#define WS_PE    294912     // 100*256 f32  = 102400
#define WS_TOTAL 397312

union FragBF { v16bf v; v8bf h[2]; };

// A-matrix 16-bit 16x32 layout (ISA 7.12.2): lanes 0-15 = M, halves of K split
// across VGPR quads; lanes 16-31 get the +8 K offset.
__device__ __forceinline__ v16bf load_fragA(const bf16* base, int row, int stride, int kb) {
  FragBF f;
  const bf16* p = base + row * stride + kb;
  f.h[0] = *(const v8bf*)(p);        // K = kb .. kb+7
  f.h[1] = *(const v8bf*)(p + 16);   // K = kb+16 .. kb+23
  return f.v;
}

// B-matrix: lane = N, same K split as A: K = kt+8h..+7 and kt+16+8h..+7 (weights stored
// transposed [N][K] so this is two 16-byte loads).
__device__ __forceinline__ v16bf load_fragB(const bf16* base, int row, int stride, int kt, int halfSel) {
  FragBF f;
  const bf16* p = base + row * stride + kt + halfSel * 8;
  f.h[0] = *(const v8bf*)(p);
  f.h[1] = *(const v8bf*)(p + 16);
  return f.v;
}

// One GEMM phase: D[112][256] (bf16, LDS) = A[112][K] (bf16, LDS) * B^T[256][K]
// (bf16, global/L2) + bias[n] (+ optional PE[m][n]).  8 waves; wave w owns
// N-tiles {2w, 2w+1} x all 7 M-tiles = 14 accumulators of v8f.
template<int K>
__device__ __forceinline__ void gemm_phase(const bf16* sA, const bf16* __restrict__ gBT,
                                           const float* __restrict__ biasVec,
                                           const float* __restrict__ pe,
                                           bf16* sD, int wave, int lane)
{
  const int halfSel = lane >> 4;
  const int lm      = lane & 15;
  const int nt0     = wave * 2;

  v8f acc[7][2];
  #pragma unroll
  for (int m = 0; m < 7; ++m) { acc[m][0] = {}; acc[m][1] = {}; }

  for (int kt = 0; kt < K; kt += 32) {
    const int kbA = kt + halfSel * 8;
    v16bf bf0 = load_fragB(gBT, nt0 * 16 + lm,       K, kt, halfSel);
    v16bf bf1 = load_fragB(gBT, (nt0 + 1) * 16 + lm, K, kt, halfSel);
    #pragma unroll
    for (int m = 0; m < 7; ++m) {
      v16bf af = load_fragA(sA, m * 16 + lm, K, kbA);
      acc[m][0] = __builtin_amdgcn_wmma_f32_16x16x32_bf16(
          false, af, false, bf0, (short)0, acc[m][0], false, false);
      acc[m][1] = __builtin_amdgcn_wmma_f32_16x16x32_bf16(
          false, af, false, bf1, (short)0, acc[m][1], false, false);
    }
  }

  // C/D layout (ISA): lane 0-15 -> N=lane, M = r (+8 for lanes 16-31).
  #pragma unroll
  for (int m = 0; m < 7; ++m)
    #pragma unroll
    for (int j = 0; j < 2; ++j) {
      const int gN = (nt0 + j) * 16 + lm;
      const float bv = biasVec[gN];
      #pragma unroll
      for (int r = 0; r < 8; ++r) {
        const int gM = m * 16 + halfSel * 8 + r;
        float val = acc[m][j][r] + bv;
        if (pe && gM < 100) val += pe[gM * 256 + gN];
        sD[gM * 256 + gN] = (gM < 100) ? (bf16)val : (bf16)0.0f;
      }
    }
}

// chain graph + self loops: e<99: (e -> e+1); e<198: (e-98 -> e-99); else self.
__device__ __forceinline__ void edge_sd(int e, int& s, int& d) {
  if (e < 99)       { s = e;       d = e + 1;  }
  else if (e < 198) { s = e - 98;  d = e - 99; }
  else              { s = e - 198; d = s;      }
}

// ---------------------------------------------------------------------------
// Prep: pack transposed bf16 weight panels + positional-encoding table in ws.
// ---------------------------------------------------------------------------
__global__ void prep_kernel(const float* __restrict__ W_exp,
                            const float* __restrict__ W_l,
                            const float* __restrict__ W_r,
                            bf16* __restrict__ wexpT, bf16* __restrict__ wlT,
                            bf16* __restrict__ wrT, float* __restrict__ pe)
{
  int idx = blockIdx.x * blockDim.x + threadIdx.x;
  if (idx < 16384) {                                   // W_exp^T [256][64]
    int n = idx >> 6, k = idx & 63;
    wexpT[idx] = (bf16)W_exp[k * 256 + n];
  } else if (idx < 81920) {                            // W_l[2]^T [256][256]
    int i = idx - 16384; int n = i >> 8, k = i & 255;
    wlT[i] = (bf16)W_l[2 * 65536 + k * 256 + n];
  } else if (idx < 147456) {                           // W_r[2]^T [256][256]
    int i = idx - 81920; int n = i >> 8, k = i & 255;
    wrT[i] = (bf16)W_r[2 * 65536 + k * 256 + n];
  } else if (idx < 173056) {                           // PE [100][256]
    int i = idx - 147456; int m = i >> 8, h = i & 255;
    int kk = h >> 1;
    float div = __expf(-(float)(2 * kk) * 9.210340371976184f / 256.0f);
    float ang = (float)m * div;
    pe[i] = (h & 1) ? __cosf(ang) : __sinf(ang);
  }
}

// ---------------------------------------------------------------------------
// Fused per-batch kernel: 2048 blocks x 256 threads (8 wave32), ~176 KB LDS.
// ---------------------------------------------------------------------------
__global__ void __launch_bounds__(256, 1)
gat_fused_kernel(const float* __restrict__ x,
                 const float* __restrict__ b_exp,
                 const float* __restrict__ b_l,
                 const float* __restrict__ b_r,
                 const float* __restrict__ att,
                 const float* __restrict__ gbias,
                 const float* __restrict__ W_fc,
                 const float* __restrict__ b_fc,
                 const bf16*  __restrict__ wexpT,
                 const bf16*  __restrict__ wlT,
                 const bf16*  __restrict__ wrT,
                 const float* __restrict__ pe,
                 float* __restrict__ out)
{
  __shared__ __align__(16) bf16 sX0[112 * 256];   // 57344 B (also fp32 x staging)
  __shared__ __align__(16) bf16 sXL[112 * 256];   // 57344 B
  __shared__ __align__(16) bf16 sXR[112 * 256];   // 57344 B (also bf16 A panel)
  __shared__ float sLogits[304];
  __shared__ float sNodeMax[128];
  __shared__ float sNodeDen[128];
  __shared__ float sAtt[256];
  __shared__ float sPool[256];

  const int tid  = threadIdx.x;
  const int wave = tid >> 5;
  const int lane = tid & 31;
  const int b    = blockIdx.x;

  // warm L2/WGP$ with the shared layer-2 weight panels
  if (tid < 64) {
    __builtin_prefetch(wlT + tid * 1024, 0, 0);
    __builtin_prefetch(wrT + tid * 1024, 0, 0);
  }

  // --- async-copy x[b] (fp32, 100*64*4 = 25600 B) straight into LDS via the
  //     CDNA5 GLOBAL_LOAD_ASYNC_TO_LDS path (no VGPR staging, ASYNCcnt) ---
  const float* xg = x + (size_t)b * (100 * 64);
  {
    const uint64_t gbase = (uint64_t)(uintptr_t)xg;
    const uint32_t lbase = (uint32_t)(uintptr_t)(void*)sX0;  // LDS byte offset
    for (int idx = tid; idx < 1600; idx += 256) {            // 1600 x 16B
      uint32_t loff = lbase + (uint32_t)(idx * 16);
      uint64_t ga   = gbase + (uint64_t)(idx * 16);
      asm volatile("global_load_async_to_lds_b128 %0, %1, off"
                   :: "v"(loff), "v"(ga) : "memory");
    }
    asm volatile("s_wait_asynccnt 0x0" ::: "memory");
  }
  sAtt[tid] = att[512 + tid];
  __syncthreads();

  // convert staged fp32 -> bf16 A panel [112][64] in the sXR region
  bf16* sXb = sXR;
  const float* xf = (const float*)sX0;
  for (int idx = tid; idx < 112 * 64; idx += 256) {
    int m = idx >> 6;
    sXb[idx] = (m < 100) ? (bf16)xf[idx] : (bf16)0.0f;
  }
  __syncthreads();

  // X0 = x @ W_exp + b_exp + PE   (overwrites the fp32 staging region)
  gemm_phase<64>(sXb, wexpT, b_exp, pe, sX0, wave, lane);
  __syncthreads();

  // XL = X0 @ W_l[2] + b_l[2];  XR = X0 @ W_r[2] + b_r[2]
  gemm_phase<256>(sX0, wlT, b_l + 512, nullptr, sXL, wave, lane);
  gemm_phase<256>(sX0, wrT, b_r + 512, nullptr, sXR, wave, lane);
  __syncthreads();

  // per-edge logits: one edge per wave iteration, 8 features per lane
  for (int e = wave; e < 298; e += 8) {
    int s, d; edge_sd(e, s, d);
    const bf16* pl = sXL + s * 256 + lane * 8;
    const bf16* pr = sXR + d * 256 + lane * 8;
    float acc = 0.0f;
    #pragma unroll
    for (int q = 0; q < 8; ++q) {
      float v = (float)pl[q] + (float)pr[q];
      v = (v > 0.0f) ? v : 0.2f * v;                 // leaky_relu(., 0.2)
      acc += sAtt[lane * 8 + q] * v;
    }
    #pragma unroll
    for (int off = 16; off > 0; off >>= 1)
      acc += __shfl_xor(acc, off, 32);
    if (lane == 0) sLogits[e] = acc;
  }
  __syncthreads();

  // per-dst softmax stats (chain graph: <=3 incoming edges, closed-form ids)
  if (tid < 100) {
    int j = tid;
    float lS = sLogits[198 + j];
    float lL = (j >= 1)  ? sLogits[j - 1]  : -3.0e38f;
    float lR = (j <= 98) ? sLogits[99 + j] : -3.0e38f;
    float mx = fmaxf(lS, fmaxf(lL, lR));
    float den = __expf(lS - mx);
    if (j >= 1)  den += __expf(lL - mx);
    if (j <= 98) den += __expf(lR - mx);
    sNodeMax[j] = mx;
    sNodeDen[j] = den;
  }
  __syncthreads();

  // pooled[h] = sum_e alpha_e * XL[src_e,h] + N * bias[2][h]
  {
    float pooled = 100.0f * gbias[512 + tid];
    for (int e = 0; e < 298; ++e) {
      int s, d; edge_sd(e, s, d);
      float alpha = __expf(sLogits[e] - sNodeMax[d]) / sNodeDen[d];
      pooled += alpha * (float)sXL[s * 256 + tid];
    }
    sPool[tid] = pooled;
  }
  __syncthreads();

  // final FC: [256] @ [256,3] + b_fc
  if (tid < 3) {
    float o = b_fc[tid];
    for (int h = 0; h < 256; ++h)
      o += sPool[h] * W_fc[h * 3 + tid];
    out[b * 3 + tid] = o;
  }
}

// ---------------------------------------------------------------------------
extern "C" void kernel_launch(void* const* d_in, const int* in_sizes, int n_in,
                              void* d_out, int out_size, void* d_ws, size_t ws_size,
                              hipStream_t stream) {
  (void)in_sizes; (void)n_in; (void)out_size; (void)ws_size;
  const float* x     = (const float*)d_in[0];
  const float* W_exp = (const float*)d_in[1];
  const float* b_exp = (const float*)d_in[2];
  const float* W_l   = (const float*)d_in[3];
  const float* b_l   = (const float*)d_in[4];
  const float* W_r   = (const float*)d_in[5];
  const float* b_r   = (const float*)d_in[6];
  const float* att   = (const float*)d_in[7];
  const float* gbias = (const float*)d_in[8];
  const float* W_fc  = (const float*)d_in[9];
  const float* b_fc  = (const float*)d_in[10];
  float* out = (float*)d_out;

  char* ws = (char*)d_ws;
  bf16*  wexpT = (bf16*)(ws + WS_WEXPT);
  bf16*  wlT   = (bf16*)(ws + WS_WLT);
  bf16*  wrT   = (bf16*)(ws + WS_WRT);
  float* pe    = (float*)(ws + WS_PE);

  prep_kernel<<<676, 256, 0, stream>>>(W_exp, W_l, W_r, wexpT, wlT, wrT, pe);
  gat_fused_kernel<<<2048, 256, 0, stream>>>(x, b_exp, b_l, b_r, att, gbias,
                                             W_fc, b_fc, wexpT, wlT, wrT, pe, out);
}
